// IDMPNN_Transformer_1065151889807
// MI455X (gfx1250) — compile-verified, matrix operand order checked by reading, not run on hardware
//
#include <hip/hip_runtime.h>

// ---------------------------------------------------------------------------
// IDMPNN Transformer attention block for gfx1250 (MI455X), bf16 WMMA path.
// B=16, N=512, D=1024, H=16, HD=64, MAX_BOND=4, MAX_DIST=37.
// ---------------------------------------------------------------------------

#define BB   16
#define NN   512
#define DD   1024
#define HH   16
#define HD   64
#define BN   (BB * NN)        // 8192 rows
#define MAXB 4
#define MAXD 37

typedef __attribute__((ext_vector_type(16))) __bf16 bf16x16;
typedef __attribute__((ext_vector_type(4)))  __bf16 bf16x4;
typedef __attribute__((ext_vector_type(8)))  float  f32x8;

// ---------------------------------------------------------------------------
// Fragment loaders.
// CDNA5 WMMA 16x16x32 bf16 per-lane layout (ISA 7.12.2): lane L (half=L>>4,
// r=L&15) holds, for a 16x32 A tile (row r) or a 32x16 B tile (column r,
// reading the *transposed* source row r), two contiguous runs of 8 K values:
//   element e<8 :  k = e     + 8*half
//   element e>=8:  k = e + 8 + 8*half
// Both runs are 16-byte contiguous in a K-major source row -> b128 loads.
// ---------------------------------------------------------------------------
__device__ inline bf16x16 load_frag_bf16(const __bf16* __restrict__ row,
                                         int kbase, int half) {
  bf16x16 f;
  const __bf16* p0 = row + kbase + half * 8;
  const __bf16* p1 = row + kbase + 16 + half * 8;
#pragma unroll
  for (int e = 0; e < 8; ++e) {
    f[e]     = p0[e];
    f[e + 8] = p1[e];
  }
  return f;
}

// ---------------------------------------------------------------------------
// fp32 -> bf16 converter
// ---------------------------------------------------------------------------
__global__ __launch_bounds__(256)
void cvt_f32_to_bf16(const float* __restrict__ src, __bf16* __restrict__ dst,
                     int n) {
  int i4 = (blockIdx.x * blockDim.x + threadIdx.x) * 4;
  if (i4 + 3 < n) {
    float4 f = *(const float4*)(src + i4);
    bf16x4 o;
    o[0] = (__bf16)f.x; o[1] = (__bf16)f.y;
    o[2] = (__bf16)f.z; o[3] = (__bf16)f.w;
    *(bf16x4*)(dst + i4) = o;
  } else {
    for (int i = i4; i < n; ++i) dst[i] = (__bf16)src[i];
  }
}

// ---------------------------------------------------------------------------
// V transpose per head: vb (B,N,D) -> vt (B,H,HD,N) so PV B-fragments are
// contiguous along the key dimension.
// ---------------------------------------------------------------------------
__global__ __launch_bounds__(256)
void transpose_v(const __bf16* __restrict__ vsrc, __bf16* __restrict__ vt) {
  int idx = blockIdx.x * blockDim.x + threadIdx.x;   // over B*N*D
  if (idx >= BB * NN * DD) return;
  int d = idx % HD;
  int h = (idx / HD) % HH;
  int j = (idx / DD) % NN;
  int b = idx / (NN * DD);
  vt[(((size_t)(b * HH + h)) * HD + d) * NN + j] = vsrc[idx];
}

// ---------------------------------------------------------------------------
// Tiled WMMA GEMM: C[M x D] = (A[M x D] @ W^T + bias) * scale.
// nn.Linear semantics: C[i,j] = dot(A[i,:], W[j,:]) -- both operands K-major.
// Wave tile = 16(M) x 64(N); 8 waves / block; K-loop step 32.
// Templated output type to keep the epilogue branch-free.
// ---------------------------------------------------------------------------
template <bool BF16OUT>
__global__ __launch_bounds__(256)
void gemm_wmma_bf16(const __bf16* __restrict__ A, const __bf16* __restrict__ W,
                    const float* __restrict__ bias, float scale,
                    void* __restrict__ outPtr, int M) {
  int wave = blockIdx.x * 8 + (threadIdx.x >> 5);
  int lane = threadIdx.x & 31;
  int half = lane >> 4, r = lane & 15;
  const int nStrips = DD / 64;                  // 16
  int mt = wave / nStrips;
  int n0 = (wave % nStrips) * 64;
  if (mt * 16 >= M) return;

  const __bf16* arow  = A + (size_t)(mt * 16 + r) * DD;
  const __bf16* wrow0 = W + (size_t)(n0 + r) * DD;   // B column r -> W row (n0+t*16+r)

  f32x8 acc[4] = {f32x8{}, f32x8{}, f32x8{}, f32x8{}};

  for (int kb = 0; kb < DD; kb += 32) {
    // prefetch one K-step ahead (L2-resident streams -> hide L2->L1 latency)
    __builtin_prefetch(arow + kb + 32, 0, 1);
#pragma unroll
    for (int t = 0; t < 4; ++t)
      __builtin_prefetch(wrow0 + (size_t)t * 16 * DD + kb + 32, 0, 1);

    bf16x16 af = load_frag_bf16(arow, kb, half);
#pragma unroll
    for (int t = 0; t < 4; ++t) {
      bf16x16 bf = load_frag_bf16(wrow0 + (size_t)t * 16 * DD, kb, half);
      acc[t] = __builtin_amdgcn_wmma_f32_16x16x32_bf16(
          false, af, false, bf, (short)0, acc[t], false, false);
    }
  }

  // C layout: VGPR v -> row (v + 8*half), lane column r.
#pragma unroll
  for (int t = 0; t < 4; ++t) {
#pragma unroll
    for (int v = 0; v < 8; ++v) {
      int row = mt * 16 + v + half * 8;
      int col = n0 + t * 16 + r;
      float val = (acc[t][v] + bias[col]) * scale;
      if (BF16OUT) ((__bf16*)outPtr)[(size_t)row * DD + col] = (__bf16)val;
      else         ((float*)outPtr)[(size_t)row * DD + col] = val;
    }
  }
}

// ---------------------------------------------------------------------------
// Flash attention with structural bias.
// Block = 512 threads = 16 waves = 16 heads, all sharing one (b, i-tile).
//
// LDS data paths:
//  - sLUT  : fused bias table  lut[(a*37+d)*16+h] = edge[a][h] + dist[d][h]
//            (9.5 KB, built once per block; one DS read replaces two)
//  - sIdx  : combined (adj*37+dist) 16x32 int tile, double-buffered so only
//            ONE __syncthreads per 32-key step is needed; adj/dist are read
//            from HBM exactly once, fully coalesced, shared by all 16 heads
//  - sMask : additive key mask (0 or -1e30) staged once per block; -1e30
//            absorbs any finite score at fp32, identical to a select
//  - sP    : per-wave 1KB slab for P relayout (C-layout -> A-layout)
// Per 32-key step and per wave: 8x v_wmma_f32_16x16x32_bf16
// (4 for QK^T over HD=64, 4 for PV over the 64 output dims).
// ---------------------------------------------------------------------------
__global__ __launch_bounds__(512)
void attn_flash_wmma(const __bf16* __restrict__ Q, const __bf16* __restrict__ K,
                     const __bf16* __restrict__ VT,
                     const int* __restrict__ adj, const int* __restrict__ dist,
                     const unsigned char* __restrict__ kpm,
                     const float* __restrict__ edge_emb,
                     const float* __restrict__ dist_emb,
                     __bf16* __restrict__ out) {
  __shared__ int    sIdx[2][16 * 32];          // double-buffered fused indices
  __shared__ float  sLUT[MAXB * MAXD * HH];    // 148 x 16 fused bias table
  __shared__ float  sMask[NN];                 // additive key-padding mask
  __shared__ __bf16 sP[16][16 * 32];           // per-wave P staging (1KB each)

  int tid  = threadIdx.x;
  int h    = tid >> 5;                         // wave index == head
  int lane = tid & 31;
  int half = lane >> 4, r = lane & 15;
  const int iTiles = NN / 16;                  // 32
  int b  = blockIdx.x / iTiles;
  int i0 = (blockIdx.x % iTiles) * 16;
  __bf16* myP = sP[h];

  // build fused bias LUT + additive mask once per block
  for (int idx = tid; idx < MAXB * MAXD * HH; idx += 512) {
    int hh = idx & (HH - 1);
    int t  = idx >> 4;                         // = a*37 + d
    int d_ = t % MAXD;
    int a_ = t / MAXD;
    sLUT[idx] = edge_emb[a_ * HH + hh] + dist_emb[d_ * HH + hh];
  }
  for (int j = tid; j < NN; j += 512)
    sMask[j] = kpm[b * NN + j] ? -1e30f : 0.0f;

  const __bf16* qrow = Q + ((size_t)(b * NN + i0 + r) * DD + h * HD);
  bf16x16 qf0 = load_frag_bf16(qrow, 0,  half);
  bf16x16 qf1 = load_frag_bf16(qrow, 32, half);

  const __bf16* vtbase = VT + ((size_t)(b * HH + h) * HD) * NN;

  // cooperative adj/dist tile coordinates: 512 threads <-> 16x32 ints
  int trow = tid >> 5;                         // 0..15 (query row)
  int tcol = tid & 31;                         // 0..31 (key col)
  const size_t adjRow = ((size_t)b * NN + (i0 + trow)) * NN;

  float m[8], l[8];
  f32x8 acc[4] = {f32x8{}, f32x8{}, f32x8{}, f32x8{}};
#pragma unroll
  for (int v = 0; v < 8; ++v) { m[v] = -1e30f; l[v] = 0.0f; }

  int buf = 0;
  for (int jt = 0; jt < NN; jt += 32, buf ^= 1) {
    // stage fused (adj*37+dist) tile: coalesced, read once for all 16 heads
    {
      int a_ = adj[adjRow + jt + tcol];
      int d_ = dist[adjRow + jt + tcol];
      sIdx[buf][trow * 32 + tcol] = a_ * MAXD + d_;
    }
    __syncthreads();   // single barrier per step (double-buffered tile)

    f32x8 s[2];
#pragma unroll
    for (int u = 0; u < 2; ++u) {
      int j0 = jt + u * 16;
      const __bf16* krow = K + ((size_t)(b * NN + j0 + r) * DD + h * HD);
      bf16x16 k0 = load_frag_bf16(krow, 0,  half);
      bf16x16 k1 = load_frag_bf16(krow, 32, half);
      f32x8 c = {};
      c = __builtin_amdgcn_wmma_f32_16x16x32_bf16(false, qf0, false, k0,
                                                  (short)0, c, false, false);
      c = __builtin_amdgcn_wmma_f32_16x16x32_bf16(false, qf1, false, k1,
                                                  (short)0, c, false, false);
      // fused structural bias + additive key mask
      int jl = u * 16 + r;
      float maskf = sMask[jt + jl];
#pragma unroll
      for (int v = 0; v < 8; ++v) {
        int il = v + half * 8;
        int fi = sIdx[buf][il * 32 + jl];
        c[v] += sLUT[fi * HH + h] + maskf;
      }
      s[u] = c;
    }

    // online softmax: per-row max/sum across the 16 lanes of each half-group
    float mnew[8], alpha[8];
#pragma unroll
    for (int v = 0; v < 8; ++v) {
      float rm = fmaxf(s[0][v], s[1][v]);
      rm = fmaxf(rm, __shfl_xor(rm, 1, 32));
      rm = fmaxf(rm, __shfl_xor(rm, 2, 32));
      rm = fmaxf(rm, __shfl_xor(rm, 4, 32));
      rm = fmaxf(rm, __shfl_xor(rm, 8, 32));
      mnew[v]  = fmaxf(m[v], rm);
      alpha[v] = __expf(m[v] - mnew[v]);
      m[v]     = mnew[v];
    }

#pragma unroll
    for (int v = 0; v < 8; ++v) {
      float p0 = __expf(s[0][v] - mnew[v]);
      float p1 = __expf(s[1][v] - mnew[v]);
      float rs = p0 + p1;
      rs += __shfl_xor(rs, 1, 32);
      rs += __shfl_xor(rs, 2, 32);
      rs += __shfl_xor(rs, 4, 32);
      rs += __shfl_xor(rs, 8, 32);
      l[v] = l[v] * alpha[v] + rs;
      int row = v + half * 8;
      myP[row * 32 + r]      = (__bf16)p0;     // keys jt..jt+15
      myP[row * 32 + 16 + r] = (__bf16)p1;     // keys jt+16..jt+31
    }

#pragma unroll
    for (int t = 0; t < 4; ++t)
#pragma unroll
      for (int v = 0; v < 8; ++v) acc[t][v] *= alpha[v];

    // P: C-layout -> A-layout via per-wave LDS slab (same-wave DS ordering)
    bf16x16 pA = load_frag_bf16(myP + r * 32, 0, half);
#pragma unroll
    for (int t = 0; t < 4; ++t) {
      bf16x16 vf = load_frag_bf16(vtbase + (size_t)(t * 16 + r) * NN, jt, half);
      acc[t] = __builtin_amdgcn_wmma_f32_16x16x32_bf16(
          false, pA, false, vf, (short)0, acc[t], false, false);
    }
  }

  // normalize and store bf16 for the output projection
#pragma unroll
  for (int t = 0; t < 4; ++t) {
#pragma unroll
    for (int v = 0; v < 8; ++v) {
      int i = i0 + v + half * 8;
      int d = h * HD + t * 16 + r;
      out[(size_t)(b * NN + i) * DD + d] = (__bf16)(acc[t][v] / l[v]);
    }
  }
}

// ---------------------------------------------------------------------------
// Host launcher
// ---------------------------------------------------------------------------
extern "C" void kernel_launch(void* const* d_in, const int* in_sizes, int n_in,
                              void* d_out, int out_size, void* d_ws,
                              size_t ws_size, hipStream_t stream) {
  const float* x        = (const float*)d_in[0];
  const int*   adj      = (const int*)d_in[1];
  const int*   dist     = (const int*)d_in[2];
  const unsigned char* kpm = (const unsigned char*)d_in[3];  // jax bool = 1 byte
  const float* Wq = (const float*)d_in[4];
  const float* bq = (const float*)d_in[5];
  const float* Wk = (const float*)d_in[6];
  const float* bk = (const float*)d_in[7];
  const float* Wv = (const float*)d_in[8];
  const float* bv = (const float*)d_in[9];
  const float* Wo = (const float*)d_in[10];
  const float* bo = (const float*)d_in[11];
  const float* edge_emb = (const float*)d_in[12];
  const float* dist_emb = (const float*)d_in[13];
  float* out = (float*)d_out;

  // workspace carve-up (bf16 intermediates): total = 104 MB
  char* ws = (char*)d_ws;
  __bf16* xb    = (__bf16*)ws; ws += (size_t)BN * DD * 2;    // 16 MB
  __bf16* wqb   = (__bf16*)ws; ws += (size_t)DD * DD * 2;    //  2 MB
  __bf16* wkb   = (__bf16*)ws; ws += (size_t)DD * DD * 2;
  __bf16* wvb   = (__bf16*)ws; ws += (size_t)DD * DD * 2;
  __bf16* wob   = (__bf16*)ws; ws += (size_t)DD * DD * 2;
  __bf16* qb    = (__bf16*)ws; ws += (size_t)BN * DD * 2;    // 16 MB
  __bf16* kb    = (__bf16*)ws; ws += (size_t)BN * DD * 2;
  __bf16* vb    = (__bf16*)ws; ws += (size_t)BN * DD * 2;
  __bf16* vt    = (__bf16*)ws; ws += (size_t)BN * DD * 2;
  __bf16* attnb = (__bf16*)ws; ws += (size_t)BN * DD * 2;

  auto cvtBlocks = [](size_t n) { return (int)((n / 4 + 255) / 256); };

  // 1) fp32 -> bf16 conversions (x read once, weights once)
  cvt_f32_to_bf16<<<cvtBlocks((size_t)BN * DD), 256, 0, stream>>>(x,  xb,  BN * DD);
  cvt_f32_to_bf16<<<cvtBlocks((size_t)DD * DD), 256, 0, stream>>>(Wq, wqb, DD * DD);
  cvt_f32_to_bf16<<<cvtBlocks((size_t)DD * DD), 256, 0, stream>>>(Wk, wkb, DD * DD);
  cvt_f32_to_bf16<<<cvtBlocks((size_t)DD * DD), 256, 0, stream>>>(Wv, wvb, DD * DD);
  cvt_f32_to_bf16<<<cvtBlocks((size_t)DD * DD), 256, 0, stream>>>(Wo, wob, DD * DD);

  // 2) projections (q folded with HD^-0.5 = 0.125)
  const int gemmBlocks = (BN / 16) * (DD / 64) / 8;   // 1024
  gemm_wmma_bf16<true><<<gemmBlocks, 256, 0, stream>>>(xb, wqb, bq, 0.125f, qb, BN);
  gemm_wmma_bf16<true><<<gemmBlocks, 256, 0, stream>>>(xb, wkb, bk, 1.0f,   kb, BN);
  gemm_wmma_bf16<true><<<gemmBlocks, 256, 0, stream>>>(xb, wvb, bv, 1.0f,   vb, BN);

  // 3) per-head V transpose for the PV contraction
  transpose_v<<<(BB * NN * DD + 255) / 256, 256, 0, stream>>>(vb, vt);

  // 4) flash attention: block = (b, i-tile), 16 head-waves share LDS tiles
  const int attnBlocks = BB * (NN / 16);              // 512
  attn_flash_wmma<<<attnBlocks, 512, 0, stream>>>(qb, kb, vt, adj, dist, kpm,
                                                  edge_emb, dist_emb, attnb);

  // 5) output projection -> fp32 d_out
  gemm_wmma_bf16<false><<<gemmBlocks, 256, 0, stream>>>(attnb, wob, bo, 1.0f,
                                                        out, BN);
}
